// TemporalAttention_16252156248499
// MI455X (gfx1250) — compile-verified, matrix operand order checked by reading, not run on hardware
//
#include <hip/hip_runtime.h>
#include <hip/hip_bf16.h>
#include <math.h>

// ---------------------------------------------------------------------------
// Problem constants (from the reference): T=512, B=64, N=1024, H=512
// ---------------------------------------------------------------------------
#define T_LEN 512
#define B_SZ  64
#define N_SZ  1024
#define H_SZ  512
#define TB    (T_LEN * B_SZ)   // 32768 flattened (t,b) rows

typedef __attribute__((ext_vector_type(16))) __bf16 v16bf;
typedef __attribute__((ext_vector_type(8)))  float  v8f;

union FragU { v16bf v; uint4 q[2]; };

// ---------------------------------------------------------------------------
// CDNA5 async LDS<->global DMA (ASYNCcnt-tracked, no VGPR data path).
// LDS byte address = low 32 bits of the flat address (ISA 10.2 aperture map).
// ---------------------------------------------------------------------------
__device__ __forceinline__ void async_ld16(__bf16* lds_dst, const __bf16* gsrc) {
  unsigned l = (unsigned)(size_t)(void*)lds_dst;
  asm volatile("global_load_async_to_lds_b128 %0, %1, off"
               :: "v"(l), "v"(gsrc) : "memory");
}
__device__ __forceinline__ void async_st16(const __bf16* gdst, const __bf16* lds_src) {
  unsigned l = (unsigned)(size_t)(void*)lds_src;
  asm volatile("global_store_async_from_lds_b128 %0, %1, off"
               :: "v"(gdst), "v"(l) : "memory");
}
__device__ __forceinline__ void wait_async0() {
  asm volatile("s_wait_asynccnt 0x0" ::: "memory");
}

// Load a 16x32 bf16 WMMA operand fragment from K-contiguous memory.
// Per CDNA5 ISA (CDNA5 ISA, 16-bit A layout): lanes 0-15 hold row M=lane with
// K=[0..8)+[16..24); lanes 16-31 hold row M=lane-16 with K=[8..16)+[24..32).
__device__ __forceinline__ v16bf load_frag(const __bf16* base, int ldk) {
  const int lane = threadIdx.x & 31;
  const int row  = lane & 15;
  const int k0   = (lane >> 4) << 3;  // 0 or 8
  const __bf16* p = base + (size_t)row * ldk + k0;
  FragU f;
  f.q[0] = *(const uint4*)(p);
  f.q[1] = *(const uint4*)(p + 16);
  return f.v;
}

__device__ __forceinline__ v8f wmma_bf16(v16bf a, v16bf b, v8f c) {
  return __builtin_amdgcn_wmma_f32_16x16x32_bf16(false, a, false, b,
                                                 (short)0, c, false, false);
}

__device__ __forceinline__ v8f vzero8() {
  v8f z = {0.f, 0.f, 0.f, 0.f, 0.f, 0.f, 0.f, 0.f};
  return z;
}

// ---------------------------------------------------------------------------
// fp32 -> bf16 conversion (grid-stride)
// ---------------------------------------------------------------------------
__global__ void cvt_f32_to_bf16(const float* __restrict__ src,
                                __bf16* __restrict__ dst, int n) {
  int i = blockIdx.x * blockDim.x + threadIdx.x;
  int stride = gridDim.x * blockDim.x;
  for (; i < n; i += stride) dst[i] = (__bf16)src[i];
}

// Extract second half of W1 ([N, 2N] fp32) into bf16 [N, N]: W1b[n][k] = W1[n][N+k]
__global__ void cvt_w1b(const float* __restrict__ W1, __bf16* __restrict__ dst) {
  int i = blockIdx.x * blockDim.x + threadIdx.x;
  int stride = gridDim.x * blockDim.x;
  const int total = N_SZ * N_SZ;
  for (; i < total; i += stride) {
    int n = i >> 10, k = i & (N_SZ - 1);
    dst[i] = (__bf16)W1[(size_t)n * (2 * N_SZ) + N_SZ + k];
  }
}

// ---------------------------------------------------------------------------
// ss[b][n] = b1[n] + s[b,:] @ W1[n, 0:N]   (time-invariant half of z @ W1^T)
// ---------------------------------------------------------------------------
__global__ void ss_kernel(const float* __restrict__ s, const float* __restrict__ W1,
                          const float* __restrict__ b1, float* __restrict__ ss) {
  int i = blockIdx.x * blockDim.x + threadIdx.x;    // 0 .. B*N-1
  int b = i >> 10, n = i & (N_SZ - 1);
  const float* srow = s + (size_t)b * N_SZ;
  const float* wrow = W1 + (size_t)n * (2 * N_SZ);
  float acc = b1[n];
  for (int k = 0; k < N_SZ; ++k) acc += srow[k] * wrow[k];
  ss[i] = acc;
}

// ---------------------------------------------------------------------------
// Input projection: xw_dir = x @ Wih_dir^T + bih_dir   (bf16 WMMA GEMM)
// Grid: (M/128, H/64, 2 directions). Block 256 = 8 waves, each wave 2x2 tiles.
// Double-buffered async global->LDS staging.
// ---------------------------------------------------------------------------
__global__ __launch_bounds__(256)
void inproj_kernel(const __bf16* __restrict__ xb,
                   const __bf16* __restrict__ Wf, const __bf16* __restrict__ Wb,
                   const float* __restrict__ bih_f, const float* __restrict__ bih_b,
                   __bf16* __restrict__ xw_f, __bf16* __restrict__ xw_b) {
  __shared__ __align__(16) __bf16 As[2][128 * 32];
  __shared__ __align__(16) __bf16 Bs[2][64 * 32];
  const int m0  = blockIdx.x * 128;
  const int n0  = blockIdx.y * 64;
  const int dir = blockIdx.z;
  const __bf16* W   = dir ? Wb : Wf;
  const float*  bih = dir ? bih_b : bih_f;
  __bf16*       out = dir ? xw_b : xw_f;

  const int tid  = threadIdx.x;
  const int wave = tid >> 5;
  const int lane = tid & 31;
  const int wy = wave & 3;   // 2 M-tiles each
  const int wx = wave >> 2;  // 2 N-tiles each

  auto stage = [&](int buf, int kk) {
#pragma unroll
    for (int c = 0; c < 2; ++c) {           // A tile 128x32: 512 16B chunks
      int chunk = tid * 2 + c;
      int r = chunk >> 2, kc = (chunk & 3) << 3;
      async_ld16(&As[buf][r * 32 + kc], xb + (size_t)(m0 + r) * N_SZ + kk + kc);
    }
    {                                        // B^T tile 64x32: 256 chunks
      int r = tid >> 2, kc = (tid & 3) << 3;
      async_ld16(&Bs[buf][r * 32 + kc], W + (size_t)(n0 + r) * N_SZ + kk + kc);
    }
  };

  v8f acc[2][2];
#pragma unroll
  for (int i = 0; i < 2; ++i)
#pragma unroll
    for (int j = 0; j < 2; ++j) acc[i][j] = vzero8();

  stage(0, 0);
  wait_async0();
  __syncthreads();

  int cur = 0;
  for (int kk = 0; kk < N_SZ; kk += 32) {
    int nxt = cur ^ 1;
    if (kk + 32 < N_SZ) stage(nxt, kk + 32);   // async prefetch next K-tile

    v16bf a0  = load_frag(&As[cur][(wy * 2 + 0) * 16 * 32], 32);
    v16bf a1  = load_frag(&As[cur][(wy * 2 + 1) * 16 * 32], 32);
    v16bf bb0 = load_frag(&Bs[cur][(wx * 2 + 0) * 16 * 32], 32);
    v16bf bb1 = load_frag(&Bs[cur][(wx * 2 + 1) * 16 * 32], 32);
    acc[0][0] = wmma_bf16(a0, bb0, acc[0][0]);
    acc[0][1] = wmma_bf16(a0, bb1, acc[0][1]);
    acc[1][0] = wmma_bf16(a1, bb0, acc[1][0]);
    acc[1][1] = wmma_bf16(a1, bb1, acc[1][1]);

    wait_async0();
    __syncthreads();
    cur = nxt;
  }

  const int rlo = (lane >> 4) << 3;
  const int cl  = lane & 15;
#pragma unroll
  for (int i = 0; i < 2; ++i)
#pragma unroll
    for (int j = 0; j < 2; ++j) {
      int mb = (wy * 2 + i) * 16, nb = (wx * 2 + j) * 16;
#pragma unroll
      for (int r = 0; r < 8; ++r) {
        int row = m0 + mb + rlo + r;
        int col = n0 + nb + cl;
        out[(size_t)row * H_SZ + col] = (__bf16)(acc[i][j][r] + bih[col]);
      }
    }
}

// ---------------------------------------------------------------------------
// Sequential bidirectional scan. grid=2 (one persistent WG per direction),
// block=512 (16 waves). LDS: h (64x512 bf16 = 64KB) + xw[t] tile (64KB).
//   - xw[t] is async-DMA'd into LDS at step start, hidden behind the GEMM
//   - h_next is written back to h_cat with async LDS->global stores
//   - Whh (512KB bf16) streams from L2 directly into fragments each step
// ---------------------------------------------------------------------------
__global__ __launch_bounds__(512)
void rnn_kernel(const __bf16* __restrict__ xw_f, const __bf16* __restrict__ xw_b,
                const __bf16* __restrict__ Whh_f, const __bf16* __restrict__ Whh_b,
                const float* __restrict__ bhh_f, const float* __restrict__ bhh_b,
                __bf16* __restrict__ h_cat) {
  extern __shared__ __align__(16) __bf16 smem[];
  __bf16* hlds  = smem;                     // [64][512]
  __bf16* xwlds = smem + B_SZ * H_SZ;       // [64][512]

  const int dir = blockIdx.x;
  const __bf16* xw  = dir ? xw_b : xw_f;
  const __bf16* Whh = dir ? Whh_b : Whh_f;
  const float*  bhh = dir ? bhh_b : bhh_f;

  const int tid = threadIdx.x, wave = tid >> 5, lane = tid & 31;
  const int rlo = (lane >> 4) << 3, cl = lane & 15;

  for (int i = tid; i < B_SZ * H_SZ; i += blockDim.x) hlds[i] = (__bf16)0.f;
  __syncthreads();

  for (int step = 0; step < T_LEN; ++step) {
    const int t = dir ? (T_LEN - 1 - step) : step;

    // async prefetch of the whole xw[t] tile (64KB) -> overlapped with GEMM
    const __bf16* xwsrc = xw + (size_t)t * B_SZ * H_SZ;
    for (int c = tid; c < (B_SZ * H_SZ) / 8; c += 512)
      async_ld16(xwlds + c * 8, xwsrc + c * 8);

    v8f acc[4][2];   // 4 M-tiles (B=64) x 2 N-tiles per wave (16 waves -> H=512)
#pragma unroll
    for (int i = 0; i < 4; ++i)
#pragma unroll
      for (int j = 0; j < 2; ++j) acc[i][j] = vzero8();

    for (int kk = 0; kk < H_SZ; kk += 32) {
      v16bf bfr[2];
#pragma unroll
      for (int j = 0; j < 2; ++j) {
        int nt = wave * 2 + j;
        bfr[j] = load_frag(Whh + (size_t)(nt * 16) * H_SZ + kk, H_SZ);  // L2-resident
      }
#pragma unroll
      for (int i = 0; i < 4; ++i) {
        v16bf afr = load_frag(hlds + (size_t)(i * 16) * H_SZ + kk, H_SZ);
        acc[i][0] = wmma_bf16(afr, bfr[0], acc[i][0]);
        acc[i][1] = wmma_bf16(afr, bfr[1], acc[i][1]);
      }
    }

    // xw tile ready + all hlds reads done + previous step's async stores
    // (which read hlds) retired -> safe to overwrite hlds.
    wait_async0();
    __syncthreads();

#pragma unroll
    for (int i = 0; i < 4; ++i)
#pragma unroll
      for (int j = 0; j < 2; ++j) {
        int nt = wave * 2 + j;
#pragma unroll
        for (int r = 0; r < 8; ++r) {
          int b   = i * 16 + rlo + r;      // batch row
          int col = nt * 16 + cl;          // hidden unit
          float v = tanhf(acc[i][j][r] + (float)xwlds[b * H_SZ + col] + bhh[col]);
          hlds[(size_t)b * H_SZ + col] = (__bf16)v;
        }
      }
    __syncthreads();   // h update visible to all waves

    // async store h tile LDS -> h_cat[t, :, dir*H .. dir*H+H)
    for (int c = tid; c < (B_SZ * H_SZ) / 8; c += 512) {
      int b = c >> 6, off = (c & 63) << 3;
      async_st16(h_cat + (size_t)(t * B_SZ + b) * N_SZ + dir * H_SZ + off,
                 hlds + b * H_SZ + off);
    }
    // no wait here: next step's wait_async0 guards the hlds WAR hazard
  }
}

// ---------------------------------------------------------------------------
// Attention scores: e[t,b] = tanh(b2 + sum_n W2[n]*tanh(ss[b,n] + h[t,b,:]@W1b[n,:]))
// Each block: 32 flattened rows x full N=1024, K=1024. Double-buffered async A.
// ---------------------------------------------------------------------------
__global__ __launch_bounds__(512)
void attn_score_kernel(const __bf16* __restrict__ h_cat, const __bf16* __restrict__ W1b,
                       const float* __restrict__ ss, const float* __restrict__ W2,
                       const float* __restrict__ b2, float* __restrict__ e) {
  __shared__ __align__(16) __bf16 As[2][32 * 32];
  __shared__ float rowacc[32];
  const int row0 = blockIdx.x * 32;
  const int tid = threadIdx.x, wave = tid >> 5, lane = tid & 31;
  const int rlo = (lane >> 4) << 3, cl = lane & 15;

  auto stage = [&](int buf, int kk) {
    if (tid < 128) {   // A tile 32x32: 128 16B chunks
      int r = tid >> 2, kc = (tid & 3) << 3;
      async_ld16(&As[buf][r * 32 + kc],
                 h_cat + (size_t)(row0 + r) * N_SZ + kk + kc);
    }
  };

  v8f acc[2][4];   // 2 M-tiles x 4 N-tiles per wave (16 waves -> N=1024)
#pragma unroll
  for (int i = 0; i < 2; ++i)
#pragma unroll
    for (int j = 0; j < 4; ++j) acc[i][j] = vzero8();

  stage(0, 0);
  wait_async0();
  __syncthreads();

  int cur = 0;
  for (int kk = 0; kk < N_SZ; kk += 32) {
    int nxt = cur ^ 1;
    if (kk + 32 < N_SZ) stage(nxt, kk + 32);

    v16bf a0 = load_frag(&As[cur][0], 32);
    v16bf a1 = load_frag(&As[cur][16 * 32], 32);
#pragma unroll
    for (int j = 0; j < 4; ++j) {
      int nt = wave * 4 + j;
      v16bf bfr = load_frag(W1b + (size_t)(nt * 16) * N_SZ + kk, N_SZ);
      acc[0][j] = wmma_bf16(a0, bfr, acc[0][j]);
      acc[1][j] = wmma_bf16(a1, bfr, acc[1][j]);
    }

    wait_async0();
    __syncthreads();
    cur = nxt;
  }

  if (tid < 32) rowacc[tid] = 0.f;
  __syncthreads();
#pragma unroll
  for (int i = 0; i < 2; ++i)
#pragma unroll
    for (int j = 0; j < 4; ++j) {
      int nt = wave * 4 + j;
#pragma unroll
      for (int r = 0; r < 8; ++r) {
        int m = i * 16 + rlo + r;
        int col = nt * 16 + cl;
        int bidx = (row0 + m) & (B_SZ - 1);
        float u = tanhf(acc[i][j][r] + ss[(size_t)bidx * N_SZ + col]);
        atomicAdd(&rowacc[m], u * W2[col]);   // ds_add_f32
      }
    }
  __syncthreads();
  if (tid < 32) e[row0 + tid] = tanhf(rowacc[tid] + b2[0]);
}

// ---------------------------------------------------------------------------
// Softmax over T per batch column. grid = B, block = T.
// ---------------------------------------------------------------------------
__global__ __launch_bounds__(512)
void softmax_kernel(const float* __restrict__ e, float* __restrict__ a) {
  __shared__ float red[T_LEN];
  const int b = blockIdx.x, t = threadIdx.x;
  float v = e[t * B_SZ + b];
  red[t] = v; __syncthreads();
  for (int s = T_LEN / 2; s > 0; s >>= 1) {
    if (t < s) red[t] = fmaxf(red[t], red[t + s]);
    __syncthreads();
  }
  float mx = red[0]; __syncthreads();
  float ex = __expf(v - mx);
  red[t] = ex; __syncthreads();
  for (int s = T_LEN / 2; s > 0; s >>= 1) {
    if (t < s) red[t] += red[t + s];
    __syncthreads();
  }
  a[t * B_SZ + b] = ex / red[0];
}

// ---------------------------------------------------------------------------
// out[b,n] = sum_t a[t,b] * h_cat[t,b,n]. grid (B, N/256), block 256.
// ---------------------------------------------------------------------------
__global__ void wsum_kernel(const float* __restrict__ a,
                            const __bf16* __restrict__ h_cat,
                            float* __restrict__ out) {
  const int b = blockIdx.x;
  const int n = blockIdx.y * 256 + threadIdx.x;
  float acc = 0.f;
  for (int t = 0; t < T_LEN; ++t)
    acc += a[t * B_SZ + b] * (float)h_cat[(size_t)(t * B_SZ + b) * N_SZ + n];
  out[(size_t)b * N_SZ + n] = acc;
}

// ---------------------------------------------------------------------------
// Launch
// ---------------------------------------------------------------------------
extern "C" void kernel_launch(void* const* d_in, const int* in_sizes, int n_in,
                              void* d_out, int out_size, void* d_ws, size_t ws_size,
                              hipStream_t stream) {
  const float* s     = (const float*)d_in[0];
  const float* x     = (const float*)d_in[1];
  const float* Wih_f = (const float*)d_in[2];
  const float* Whh_f = (const float*)d_in[3];
  const float* bih_f = (const float*)d_in[4];
  const float* bhh_f = (const float*)d_in[5];
  const float* Wih_b = (const float*)d_in[6];
  const float* Whh_b = (const float*)d_in[7];
  const float* bih_b = (const float*)d_in[8];
  const float* bhh_b = (const float*)d_in[9];
  const float* W1    = (const float*)d_in[10];
  const float* b1    = (const float*)d_in[11];
  const float* W2    = (const float*)d_in[12];
  const float* b2    = (const float*)d_in[13];
  float* out = (float*)d_out;

  // workspace carve-up (all offsets 256B-aligned)
  char* ws = (char*)d_ws;
  size_t o = 0;
  auto carve = [&](size_t bytes) {
    char* p = ws + o;
    o += (bytes + 255) & ~(size_t)255;
    return p;
  };
  __bf16* xb    = (__bf16*)carve((size_t)TB * N_SZ * 2);     // 64 MB
  __bf16* Wfb   = (__bf16*)carve((size_t)H_SZ * N_SZ * 2);   // 1 MB
  __bf16* Wbb   = (__bf16*)carve((size_t)H_SZ * N_SZ * 2);
  __bf16* Whfb  = (__bf16*)carve((size_t)H_SZ * H_SZ * 2);   // 512 KB
  __bf16* Whbb  = (__bf16*)carve((size_t)H_SZ * H_SZ * 2);
  __bf16* W1bb  = (__bf16*)carve((size_t)N_SZ * N_SZ * 2);   // 2 MB
  __bf16* xw_f  = (__bf16*)carve((size_t)TB * H_SZ * 2);     // 32 MB
  __bf16* xw_b  = (__bf16*)carve((size_t)TB * H_SZ * 2);
  __bf16* h_cat = (__bf16*)carve((size_t)TB * N_SZ * 2);     // 64 MB
  float*  ssb   = (float*)carve((size_t)B_SZ * N_SZ * 4);
  float*  ebuf  = (float*)carve((size_t)TB * 4);
  float*  abuf  = (float*)carve((size_t)TB * 4);
  (void)ws_size; (void)in_sizes; (void)n_in; (void)out_size;

  // 1) bf16 conversions
  cvt_f32_to_bf16<<<4096, 256, 0, stream>>>(x, xb, TB * N_SZ);
  cvt_f32_to_bf16<<<512, 256, 0, stream>>>(Wih_f, Wfb, H_SZ * N_SZ);
  cvt_f32_to_bf16<<<512, 256, 0, stream>>>(Wih_b, Wbb, H_SZ * N_SZ);
  cvt_f32_to_bf16<<<256, 256, 0, stream>>>(Whh_f, Whfb, H_SZ * H_SZ);
  cvt_f32_to_bf16<<<256, 256, 0, stream>>>(Whh_b, Whbb, H_SZ * H_SZ);
  cvt_w1b<<<1024, 256, 0, stream>>>(W1, W1bb);

  // 2) time-invariant half of the attention MLP
  ss_kernel<<<(B_SZ * N_SZ) / 256, 256, 0, stream>>>(s, W1, b1, ssb);

  // 3) input projections (both directions), WMMA + async staging
  inproj_kernel<<<dim3(TB / 128, H_SZ / 64, 2), 256, 0, stream>>>(
      xb, Wfb, Wbb, bih_f, bih_b, xw_f, xw_b);

  // 4) sequential bidirectional scan, one persistent WG per direction
  //    dynamic LDS: h state (64KB) + xw[t] staging tile (64KB)
  rnn_kernel<<<2, 512, 2 * B_SZ * H_SZ * sizeof(__bf16), stream>>>(
      xw_f, xw_b, Whfb, Whbb, bhh_f, bhh_b, h_cat);

  // 5) attention scores (WMMA GEMM + fused tanh/dot epilogue)
  attn_score_kernel<<<TB / 32, 512, 0, stream>>>(h_cat, W1bb, ssb, W2, b2, ebuf);

  // 6) softmax over time, 7) weighted sum
  softmax_kernel<<<B_SZ, T_LEN, 0, stream>>>(ebuf, abuf);
  wsum_kernel<<<dim3(B_SZ, N_SZ / 256), 256, 0, stream>>>(abuf, h_cat, out);
}
